// nnue_56977036149139
// MI455X (gfx1250) — matrix operand order repaired and checked by reading, not compiled, not run on hardware
//
#include <hip/hip_runtime.h>
#include <stdint.h>

typedef __attribute__((ext_vector_type(16))) _Float16 v16h;
typedef __attribute__((ext_vector_type(8)))  _Float16 v8h;
typedef __attribute__((ext_vector_type(8)))  float    v8f;
typedef __attribute__((ext_vector_type(4)))  float    v4f;
typedef int v4i __attribute__((vector_size(16)));      // GCC-vector spelling to match builtin signature

#define FT_IN   41024
#define K_HALF  256
#define FT_OUT  512
#define BATCH   8192
#define KC      32                 // WMMA f16 K depth per step
#define NCHUNK  (FT_IN / KC)       // 1282, exact
#define LDSB    40                 // halfs per B row in LDS (32 data + 8 pad, 80B stride: 16B aligned, conflict-free)
#define ROWS_WG 64                 // waves 0-3: rows x cols[0,128); waves 4-7: rows x cols[128,256)

#if defined(__HIP_DEVICE_COMPILE__) && __has_builtin(__builtin_amdgcn_global_load_async_to_lds_b128)
#define HAVE_ASYNC_LDS 1
#else
#define HAVE_ASYNC_LDS 0
#endif

// ---------------- kernel 1: ft_w fp32 -> f16 (21 MB into ws) ----------------
__global__ __launch_bounds__(256) void wconv_kernel(const float* __restrict__ w,
                                                    _Float16* __restrict__ w16) {
    int i = (blockIdx.x * 256 + threadIdx.x) * 8;   // total = 256*41024, divisible by 8*256
    v4f a = __builtin_nontemporal_load((const v4f*)(w + i));
    v4f b = __builtin_nontemporal_load((const v4f*)(w + i + 4));
    v8h o;
#pragma unroll
    for (int j = 0; j < 4; ++j) { o[j] = (_Float16)a[j]; o[4 + j] = (_Float16)b[j]; }
    *(v8h*)(w16 + i) = o;
}

// ---------------- B staging: 256 threads stage 256 rows x 32 K f16 ----------
__device__ __forceinline__ void stageB(const _Float16* __restrict__ w16,
                                       _Float16* __restrict__ bs, int k0) {
    const int n = threadIdx.x;
    const _Float16* src = w16 + (size_t)n * FT_IN + k0;
    _Float16* dst = bs + n * LDSB;
#if HAVE_ASYNC_LDS
    // CDNA5 async copy: global -> LDS without VGPR round trip (ASYNCcnt tracked)
    typedef __attribute__((address_space(1))) v4i* gp_t;   // global int4*
    typedef __attribute__((address_space(3))) v4i* lp_t;   // LDS int4*
    gp_t g = (gp_t)(unsigned long long)(uintptr_t)src;
    lp_t l = (lp_t)(unsigned int)(uintptr_t)dst;
    __builtin_amdgcn_global_load_async_to_lds_b128(g + 0, l + 0, 0, 0);
    __builtin_amdgcn_global_load_async_to_lds_b128(g + 1, l + 1, 0, 0);
    __builtin_amdgcn_global_load_async_to_lds_b128(g + 2, l + 2, 0, 0);
    __builtin_amdgcn_global_load_async_to_lds_b128(g + 3, l + 3, 0, 0);
#else
    const v8h* s = (const v8h*)src;
    v8h d0 = s[0], d1 = s[1], d2 = s[2], d3 = s[3];
    *(v8h*)(dst + 0)  = d0;
    *(v8h*)(dst + 8)  = d1;
    *(v8h*)(dst + 16) = d2;
    *(v8h*)(dst + 24) = d3;
#endif
}

// Precise workgroup barrier for the LDS double-buffer handoff: waits only on
// DScnt (our B-fragment reads) + ASYNCcnt (our async staging writes), then
// split signal/wait. Crucially does NOT drain LOADcnt, so the A-feature
// prefetch (HBM latency) stays in flight across the chunk boundary.
__device__ __forceinline__ void wg_barrier_lds() {
    asm volatile("s_wait_dscnt 0\n\t"
                 "s_wait_asynccnt 0\n\t"
                 "s_barrier_signal -1\n\t"
                 "s_barrier_wait -1" ::: "memory");
}

// pin the pipeline: all 16 ds_load_b128 (B fragments) issue BEFORE the 8 WMMAs
__device__ __forceinline__ void sched_pin_ds_then_wmma() {
#if __has_builtin(__builtin_amdgcn_sched_group_barrier)
    __builtin_amdgcn_sched_group_barrier(0x100, 16, 0);  // 16 x DS read
    __builtin_amdgcn_sched_group_barrier(0x008, 8, 0);   // 8 x WMMA (MFMA class)
#endif
}

// ---------------- B fragment load from LDS ----------------------------------
__device__ __forceinline__ v16h ldB(const _Float16* __restrict__ bb, int n, int khi) {
    const _Float16* bp = bb + n * LDSB + khi * 8;
    v8h b0 = *(const v8h*)bp, b1 = *(const v8h*)(bp + 16);
    return __builtin_shufflevector(b0, b1, 0, 1, 2, 3, 4, 5, 6, 7,
                                           8, 9, 10, 11, 12, 13, 14, 15);
}

// A raw prefetch: features are stream-once -> non-temporal (protect L2-resident weights)
__device__ __forceinline__ v4f ldgnt(const float* p) {
    return __builtin_nontemporal_load((const v4f*)p);
}

// ---------------- kernel 2: feature transform GEMM via WMMA -----------------
__global__ __launch_bounds__(256)
__attribute__((amdgpu_waves_per_eu(2)))            // true residency: 8 waves / 4 SIMDs = 2 per EU
void ft_gemm_kernel(const float* __restrict__ f1,
                    const float* __restrict__ f2,
                    const _Float16* __restrict__ w16,
                    const float* __restrict__ ft_b,
                    _Float16* __restrict__ ftout) {
    __shared__ _Float16 bs[2][K_HALF * LDSB];      // 2 x 20 KB double buffer

    const int persp = blockIdx.y;
    const float* __restrict__ feat = persp ? f2 : f1;
    const int row0 = blockIdx.x * ROWS_WG;
    const int wave = threadIdx.x >> 5;
    const int lane = threadIdx.x & 31;
    const int m16  = lane & 15;                    // A row within tile / C column
    const int khi  = lane >> 4;                    // K-half selector (ISA 16-bit A/B layouts)
    const int rw   = wave & 3;                     // row-tile within WG
    const int cg   = wave >> 2;                    // column half: 0 -> cols 0-127, 1 -> cols 128-255
    const int nb   = cg * 128 + m16;               // base LDS row (output column) for this lane

    const float* __restrict__ arow = feat + (size_t)(row0 + rw * 16 + m16) * FT_IN;

    v8f acc[8] = {};                               // 8 N-tiles x 8 VGPRs

    // prologue: stage B chunk 0, issue A chunk-0 raw loads (converted in iter 0)
    stageB(w16, bs[0], 0);
    v4f p0 = ldgnt(arow + khi * 8);
    v4f p1 = ldgnt(arow + khi * 8 + 4);
    v4f p2 = ldgnt(arow + 16 + khi * 8);
    v4f p3 = ldgnt(arow + 16 + khi * 8 + 4);
    wg_barrier_lds();

    // steady state: branch-free body = one scheduling region
#pragma unroll 1
    for (int c = 0; c < NCHUNK - 1; ++c) {
        const int kn = (c + 1) * KC;

        // issue next-chunk work first: B staging (async->LDS) + raw A prefetch
        stageB(w16, bs[(c + 1) & 1], kn);
        v4f q0 = ldgnt(arow + kn + khi * 8);
        v4f q1 = ldgnt(arow + kn + khi * 8 + 4);
        v4f q2 = ldgnt(arow + kn + 16 + khi * 8);
        v4f q3 = ldgnt(arow + kn + 16 + khi * 8 + 4);

        // convert THIS chunk's A (its loads have been in flight a full chunk;
        // only s_wait_loadcnt<=4 needed since 4 newer loads are outstanding)
        v16h a;
#pragma unroll
        for (int j = 0; j < 4; ++j) {
            a[j]      = (_Float16)p0[j];
            a[4 + j]  = (_Float16)p1[j];
            a[8 + j]  = (_Float16)p2[j];
            a[12 + j] = (_Float16)p3[j];
        }

        // preload ALL 8 B fragments, then run the 8 WMMAs back-to-back
        const _Float16* __restrict__ bb = bs[c & 1];
        v16h bf[8];
#pragma unroll
        for (int t = 0; t < 8; ++t) bf[t] = ldB(bb, nb + t * 16, khi);
#pragma unroll
        for (int t = 0; t < 8; ++t) {
            acc[t] = __builtin_amdgcn_wmma_f32_16x16x32_f16(
                false, a, false, bf[t], (short)0, acc[t], false, false);
        }
        sched_pin_ds_then_wmma();

        wg_barrier_lds();       // DS/ASYNC-only barrier; A loads stay in flight
        p0 = q0; p1 = q1; p2 = q2; p3 = q3;
    }

    // final chunk (no staging / prefetch)
    {
        v16h a;
#pragma unroll
        for (int j = 0; j < 4; ++j) {
            a[j]      = (_Float16)p0[j];
            a[4 + j]  = (_Float16)p1[j];
            a[8 + j]  = (_Float16)p2[j];
            a[12 + j] = (_Float16)p3[j];
        }
        const _Float16* __restrict__ bb = bs[(NCHUNK - 1) & 1];
        v16h bf[8];
#pragma unroll
        for (int t = 0; t < 8; ++t) bf[t] = ldB(bb, nb + t * 16, khi);
#pragma unroll
        for (int t = 0; t < 8; ++t) {
            acc[t] = __builtin_amdgcn_wmma_f32_16x16x32_f16(
                false, a, false, bf[t], (short)0, acc[t], false, false);
        }
        sched_pin_ds_then_wmma();
    }

    // Epilogue: C layout -> row M = khi*8 + r, col N = m16; add bias, clip01, store f16
#pragma unroll
    for (int t = 0; t < 8; ++t) {
        const int col  = cg * 128 + t * 16 + m16;
        const float bv = ft_b[col];
#pragma unroll
        for (int r = 0; r < 8; ++r) {
            const int row = row0 + rw * 16 + khi * 8 + r;
            float v = acc[t][r] + bv;
            v = v < 0.0f ? 0.0f : (v > 1.0f ? 1.0f : v);
            ftout[(size_t)row * FT_OUT + (size_t)persp * K_HALF + col] = (_Float16)v;
        }
    }
}

// ---------------- kernel 3: tiny fused tail (h1 -> h2 -> out) ---------------
__global__ __launch_bounds__(256) void tail_kernel(const _Float16* __restrict__ ftout,
                                                   const float* __restrict__ h1_w,
                                                   const float* __restrict__ h1_b,
                                                   const float* __restrict__ h2_w,
                                                   const float* __restrict__ h2_b,
                                                   const float* __restrict__ out_w,
                                                   const float* __restrict__ out_b,
                                                   float* __restrict__ out) {
    __shared__ float h1t[FT_OUT * 32];             // transposed: [j][i], 64 KB
    __shared__ float h2s[32 * 32];

    for (int idx = threadIdx.x; idx < FT_OUT * 32; idx += 256) {
        int i = idx & 31, j = idx >> 5;
        h1t[idx] = h1_w[i * FT_OUT + j];
    }
    for (int idx = threadIdx.x; idx < 32 * 32; idx += 256) h2s[idx] = h2_w[idx];
    __syncthreads();

    const int row = blockIdx.x * 256 + threadIdx.x;
    const _Float16* __restrict__ xr = ftout + (size_t)row * FT_OUT;

    float a1[32];
#pragma unroll
    for (int i = 0; i < 32; ++i) a1[i] = h1_b[i];

#pragma unroll 1
    for (int j = 0; j < FT_OUT; j += 8) {
        v8h xv = *(const v8h*)(xr + j);
#pragma unroll
        for (int jj = 0; jj < 8; ++jj) {
            const float x = (float)xv[jj];
            const float* wj = h1t + (j + jj) * 32;
#pragma unroll
            for (int i = 0; i < 32; ++i) a1[i] = fmaf(x, wj[i], a1[i]);
        }
    }
#pragma unroll
    for (int i = 0; i < 32; ++i) {
        float v = a1[i];
        a1[i] = v < 0.0f ? 0.0f : (v > 1.0f ? 1.0f : v);
    }

    float a2[32];
#pragma unroll
    for (int i2 = 0; i2 < 32; ++i2) {
        float v = h2_b[i2];
        const float* wr = h2s + i2 * 32;
#pragma unroll
        for (int i = 0; i < 32; ++i) v = fmaf(a1[i], wr[i], v);
        a2[i2] = v < 0.0f ? 0.0f : (v > 1.0f ? 1.0f : v);
    }

    float o = out_b[0];
#pragma unroll
    for (int i = 0; i < 32; ++i) o = fmaf(a2[i], out_w[i], o);
    out[row] = o;
}

// ---------------------------- launcher --------------------------------------
extern "C" void kernel_launch(void* const* d_in, const int* in_sizes, int n_in,
                              void* d_out, int out_size, void* d_ws, size_t ws_size,
                              hipStream_t stream) {
    const float* f1    = (const float*)d_in[0];
    const float* f2    = (const float*)d_in[1];
    const float* ft_w  = (const float*)d_in[2];
    const float* ft_b  = (const float*)d_in[3];
    const float* h1_w  = (const float*)d_in[4];
    const float* h1_b  = (const float*)d_in[5];
    const float* h2_w  = (const float*)d_in[6];
    const float* h2_b  = (const float*)d_in[7];
    const float* out_w = (const float*)d_in[8];
    const float* out_b = (const float*)d_in[9];
    float* out = (float*)d_out;

    const size_t w16_bytes = (size_t)K_HALF * FT_IN * sizeof(_Float16);   // 21,004,288
    _Float16* w16   = (_Float16*)d_ws;
    _Float16* ftout = (_Float16*)((char*)d_ws + w16_bytes);               // + 8,388,608

    // 1) weights fp32 -> f16
    wconv_kernel<<<(K_HALF * FT_IN) / (256 * 8), 256, 0, stream>>>(ft_w, w16);

    // 2) feature-transform GEMM (both perspectives), clip01, f16 out
    dim3 grid(BATCH / ROWS_WG, 2, 1);
    ft_gemm_kernel<<<grid, 256, 0, stream>>>(f1, f2, w16, ft_b, ftout);

    // 3) tail MLP
    tail_kernel<<<BATCH / 256, 256, 0, stream>>>(ftout, h1_w, h1_b, h2_w, h2_b,
                                                 out_w, out_b, out);
}
